// FluxonUpdater_15444702396963
// MI455X (gfx1250) — compile-verified, matrix operand order checked by reading, not run on hardware
//
#include <hip/hip_runtime.h>

// ---------------------------------------------------------------------------
// FluxonUpdater for MI455X (gfx1250, wave32, WMMA)
//   B=65536, K_ROUTE=4, D=256, K_TOTAL=8192, IN_DIM=512, EMA_M=0.5
//
// Pipeline:
//   count/scan/fill : counting-sort contributions into per-fluxon buckets
//                     (integer atomics only; kills the 67M f32-atomic scatter)
//   gemm1           : m = [h_fast|h_slow] @ W_m^T   (WMMA bf16, f32 acc)
//   gather          : agg_mean[k] = sum_w(m rows) / wsum  (streaming, 1 wave/fluxon)
//   gru             : fused GRUCell + used-mask + EMA blend (WMMA bf16)
// ---------------------------------------------------------------------------

typedef __attribute__((ext_vector_type(16))) __bf16 v16bf;
typedef __attribute__((ext_vector_type(8)))  float  v8f;

#if defined(__gfx1250__) && __has_builtin(__builtin_amdgcn_global_load_async_to_lds_b128)
#define HAVE_ASYNC_LDS 1
typedef int v4i_async __attribute__((ext_vector_type(4)));
typedef __attribute__((address_space(1))) v4i_async* gptr_v4i;   // global b128
typedef __attribute__((address_space(3))) v4i_async* lptr_v4i;   // LDS b128
#endif

namespace cfg {
constexpr int BATCH   = 65536;
constexpr int KROUTE  = 4;
constexpr int DIM     = 256;
constexpr int KTOT    = 8192;
constexpr int INDIM   = 512;
constexpr float EMA_M = 0.5f;
constexpr float EPS   = 1e-9f;
}

// round-to-nearest f32 -> bf16, packed pair into one dword
__device__ __forceinline__ unsigned bf16x2(float a, float b) {
  unsigned ua = __float_as_uint(a);
  unsigned ub = __float_as_uint(b);
  ua = (ua + 0x7FFFu + ((ua >> 16) & 1u)) >> 16;
  ub = (ub + 0x7FFFu + ((ub >> 16) & 1u)) >> 16;
  return ua | (ub << 16);
}

// K-offset (even) for fragment dword p (0..7) given lane half, per ISA 7.12.2
__device__ __forceinline__ int frag_koff(int p, int half) {
  return ((p < 4) ? (p << 1) : (16 + ((p - 4) << 1))) + (half << 3);
}

// ---------------------------------------------------------------------------
__global__ void k_zero_i32(int* __restrict__ p, int n) {
  int i = blockIdx.x * blockDim.x + threadIdx.x;
  if (i < n) p[i] = 0;
}

__global__ void k_cvt(const float* __restrict__ src, unsigned* __restrict__ dst,
                      int npairs) {
  int i = blockIdx.x * blockDim.x + threadIdx.x;
  if (i < npairs) dst[i] = bf16x2(src[2 * i], src[2 * i + 1]);
}

// ---------------------------------------------------------------------------
// counting sort of the 262144 (sample, weight) contributions by fluxon id
// ---------------------------------------------------------------------------
__global__ void k_count(const int* __restrict__ idx, int* __restrict__ counts) {
  int i = blockIdx.x * blockDim.x + threadIdx.x;
  if (i < cfg::BATCH * cfg::KROUTE) atomicAdd(&counts[idx[i]], 1);
}

// single block, 256 threads: exclusive scan of counts[8192] -> offsets[8193]
__global__ void __launch_bounds__(256)
k_scan(const int* __restrict__ counts, int* __restrict__ offsets) {
  __shared__ int partials[256];
  const int t = threadIdx.x;
  int vals[32];
  int local = 0;
  #pragma unroll
  for (int i = 0; i < 32; ++i) { vals[i] = counts[t * 32 + i]; local += vals[i]; }
  partials[t] = local;
  __syncthreads();
  for (int s = 1; s < 256; s <<= 1) {
    int add = (t >= s) ? partials[t - s] : 0;
    __syncthreads();
    partials[t] += add;
    __syncthreads();
  }
  int prefix = (t == 0) ? 0 : partials[t - 1];
  #pragma unroll
  for (int i = 0; i < 32; ++i) {
    offsets[t * 32 + i] = prefix;
    prefix += vals[i];
  }
  if (t == 255) offsets[8192] = prefix;
}

__global__ void k_fill(const int* __restrict__ idx, const float* __restrict__ w,
                       const int* __restrict__ offsets, int* __restrict__ cursor,
                       int* __restrict__ entry_b, float* __restrict__ entry_w) {
  int i = blockIdx.x * blockDim.x + threadIdx.x;
  if (i >= cfg::BATCH * cfg::KROUTE) return;
  int k = idx[i];
  int pos = offsets[k] + atomicAdd(&cursor[k], 1);
  entry_b[pos] = i >> 2;   // sample row
  entry_w[pos] = w[i];
}

// ---------------------------------------------------------------------------
// GEMM1: m = [h_fast|h_slow] @ W_m^T  (f32 out, no scatter)
// block = 256 thr = 8 waves; blockIdx.x in [0,8192):
//   rt = bx>>1 (16 batch rows), ct = (bx&1)*8 + wave (16 D-cols each)
// ---------------------------------------------------------------------------
__global__ void __launch_bounds__(256)
k_gemm1(const float* __restrict__ hf, const float* __restrict__ hs,
        const unsigned* __restrict__ wm_bf,  // W_m bf16 pairs [256][256]
        float* __restrict__ mout) {
  __shared__ unsigned lds_x[16 * 256];  // 16 rows x 512 bf16 (16 KB)

  const int t    = threadIdx.x;
  const int wid  = t >> 5;
  const int lane = t & 31;
  const int rt   = blockIdx.x >> 1;
  const int ct   = ((blockIdx.x & 1) << 3) + wid;

  const float2* hf2 = (const float2*)hf;
  const float2* hs2 = (const float2*)hs;

  #pragma unroll
  for (int i = 0; i < 16; ++i) {
    int f   = i * 256 + t;          // 4096 dwords total
    int row = f >> 8;
    int kp  = f & 255;              // pair index within 512-wide row
    int b   = rt * 16 + row;
    float2 v = (kp < 128) ? hf2[b * 128 + kp] : hs2[b * 128 + kp - 128];
    lds_x[f] = bf16x2(v.x, v.y);
  }
  __syncthreads();

  const int half = lane >> 4;
  const int mn   = lane & 15;
  const unsigned* wrow = wm_bf + (ct * 16 + mn) * 256;
  __builtin_prefetch(wrow, 0, 0);

  v8f acc = {};
  #pragma unroll
  for (int kt = 0; kt < 16; ++kt) {          // K = 512 = 16 x 32
    union { v16bf v; unsigned u[8]; } A, Bm;
    #pragma unroll
    for (int p = 0; p < 8; ++p) {
      int k = kt * 32 + frag_koff(p, half);
      A.u[p]  = lds_x[mn * 256 + (k >> 1)];
      Bm.u[p] = wrow[k >> 1];
    }
    acc = __builtin_amdgcn_wmma_f32_16x16x32_bf16(
        false, A.v, false, Bm.v, (short)0, acc, false, false);
  }

  const int dcol = ct * 16 + mn;
  #pragma unroll
  for (int r = 0; r < 8; ++r) {
    int mrow = rt * 16 + r + (half << 3);
    mout[(size_t)mrow * 256 + dcol] = acc[r];
  }
}

// ---------------------------------------------------------------------------
// Bucketed gather: 1 wave per fluxon. agg_mean -> bf16, wsum -> f32.
// ---------------------------------------------------------------------------
__global__ void __launch_bounds__(256)
k_gather(const float* __restrict__ m, const int* __restrict__ offsets,
         const int* __restrict__ entry_b, const float* __restrict__ entry_w,
         float* __restrict__ wsum, unsigned* __restrict__ am_bf) {
  const int wid  = threadIdx.x >> 5;
  const int lane = threadIdx.x & 31;
  const int k    = blockIdx.x * 8 + wid;          // fluxon id

  const int base = offsets[k];
  const int end  = offsets[k + 1];

  float a0x = 0.f, a0y = 0.f, a0z = 0.f, a0w = 0.f;
  float a1x = 0.f, a1y = 0.f, a1z = 0.f, a1w = 0.f;
  float wacc = 0.f;

  for (int e = base; e < end; ++e) {
    int   b = entry_b[e];
    float w = entry_w[e];
    wacc += w;
    const float4* row = (const float4*)(m + (size_t)b * 256);
    float4 x0 = row[lane];        // cols lane*4 .. lane*4+3
    float4 x1 = row[lane + 32];   // cols 128+lane*4 ..
    a0x += w * x0.x; a0y += w * x0.y; a0z += w * x0.z; a0w += w * x0.w;
    a1x += w * x1.x; a1y += w * x1.y; a1z += w * x1.z; a1w += w * x1.w;
  }

  float inv = 1.0f / (wacc + cfg::EPS);
  unsigned* dst = am_bf + (size_t)k * 128;
  dst[lane * 2]          = bf16x2(a0x * inv, a0y * inv);
  dst[lane * 2 + 1]      = bf16x2(a0z * inv, a0w * inv);
  dst[64 + lane * 2]     = bf16x2(a1x * inv, a1y * inv);
  dst[64 + lane * 2 + 1] = bf16x2(a1z * inv, a1w * inv);
  if (lane == 0) wsum[k] = wacc;
}

// ---------------------------------------------------------------------------
// Fused GRU cell + used-mask + EMA blend (WMMA bf16, 6 accumulators/wave)
// ---------------------------------------------------------------------------
__global__ void __launch_bounds__(256)
k_gru(const unsigned* __restrict__ am_bf, const unsigned* __restrict__ st_bf,
      const unsigned* __restrict__ wih_bf, const unsigned* __restrict__ whh_bf,
      const float* __restrict__ b_ih, const float* __restrict__ b_hh,
      const float* __restrict__ states, const float* __restrict__ wsum,
      float* __restrict__ out) {
  __shared__ unsigned lds_a[16 * 128];  // agg_mean rows (bf16 pairs, 8 KB)
  __shared__ unsigned lds_s[16 * 128];  // old state rows

  const int t    = threadIdx.x;
  const int wid  = t >> 5;
  const int lane = t & 31;
  const int rt   = blockIdx.x >> 1;
  const int ct   = ((blockIdx.x & 1) << 3) + wid;

#if HAVE_ASYNC_LDS
  // pure global->LDS memcpy: use CDNA5 async-to-LDS DMA path (ASYNCcnt)
  #pragma unroll
  for (int i = 0; i < 2; ++i) {
    int c = i * 256 + t;  // 512 x 16B chunks per 8 KB buffer
    __builtin_amdgcn_global_load_async_to_lds_b128(
        (gptr_v4i)(am_bf + (size_t)rt * 2048 + c * 4),
        (lptr_v4i)(&lds_a[c * 4]), 0, 0);
    __builtin_amdgcn_global_load_async_to_lds_b128(
        (gptr_v4i)(st_bf + (size_t)rt * 2048 + c * 4),
        (lptr_v4i)(&lds_s[c * 4]), 0, 0);
  }
#if __has_builtin(__builtin_amdgcn_s_wait_asynccnt)
  __builtin_amdgcn_s_wait_asynccnt(0);
#else
  asm volatile("s_wait_asynccnt 0x0" ::: "memory");
#endif
#else
  #pragma unroll
  for (int i = 0; i < 8; ++i) {
    int f = i * 256 + t;              // 2048 dwords each
    lds_a[f] = am_bf[(size_t)rt * 2048 + f];
    lds_s[f] = st_bf[(size_t)rt * 2048 + f];
  }
#endif
  __syncthreads();

  const int half = lane >> 4;
  const int mn   = lane & 15;
  const int dcol = ct * 16 + mn;

  v8f gi0 = {}, gi1 = {}, gi2 = {};
  v8f gh0 = {}, gh1 = {}, gh2 = {};

  #pragma unroll
  for (int kt = 0; kt < 8; ++kt) {     // K = 256 = 8 x 32
    union { v16bf v; unsigned u[8]; } A, S, Bi0, Bi1, Bi2, Bh0, Bh1, Bh2;
    #pragma unroll
    for (int p = 0; p < 8; ++p) {
      int k  = kt * 32 + frag_koff(p, half);
      int kp = k >> 1;
      A.u[p]   = lds_a[mn * 128 + kp];
      S.u[p]   = lds_s[mn * 128 + kp];
      Bi0.u[p] = wih_bf[(0 * 256 + dcol) * 128 + kp];
      Bi1.u[p] = wih_bf[(1 * 256 + dcol) * 128 + kp];
      Bi2.u[p] = wih_bf[(2 * 256 + dcol) * 128 + kp];
      Bh0.u[p] = whh_bf[(0 * 256 + dcol) * 128 + kp];
      Bh1.u[p] = whh_bf[(1 * 256 + dcol) * 128 + kp];
      Bh2.u[p] = whh_bf[(2 * 256 + dcol) * 128 + kp];
    }
    gi0 = __builtin_amdgcn_wmma_f32_16x16x32_bf16(false, A.v, false, Bi0.v, (short)0, gi0, false, false);
    gi1 = __builtin_amdgcn_wmma_f32_16x16x32_bf16(false, A.v, false, Bi1.v, (short)0, gi1, false, false);
    gi2 = __builtin_amdgcn_wmma_f32_16x16x32_bf16(false, A.v, false, Bi2.v, (short)0, gi2, false, false);
    gh0 = __builtin_amdgcn_wmma_f32_16x16x32_bf16(false, S.v, false, Bh0.v, (short)0, gh0, false, false);
    gh1 = __builtin_amdgcn_wmma_f32_16x16x32_bf16(false, S.v, false, Bh1.v, (short)0, gh1, false, false);
    gh2 = __builtin_amdgcn_wmma_f32_16x16x32_bf16(false, S.v, false, Bh2.v, (short)0, gh2, false, false);
  }

  const float bi_r = b_ih[dcol], bi_z = b_ih[256 + dcol], bi_n = b_ih[512 + dcol];
  const float bh_r = b_hh[dcol], bh_z = b_hh[256 + dcol], bh_n = b_hh[512 + dcol];

  #pragma unroll
  for (int r = 0; r < 8; ++r) {
    int   m    = r + (half << 3);
    int   kk   = rt * 16 + m;
    float hold = states[(size_t)kk * 256 + dcol];

    float i_r = gi0[r] + bi_r, i_z = gi1[r] + bi_z, i_n = gi2[r] + bi_n;
    float h_r = gh0[r] + bh_r, h_z = gh1[r] + bh_z, h_n = gh2[r] + bh_n;

    float rg = 1.0f / (1.0f + __expf(-(i_r + h_r)));
    float zg = 1.0f / (1.0f + __expf(-(i_z + h_z)));
    float ng = tanhf(i_n + rg * h_n);
    float nh = (1.0f - zg) * ng + zg * hold;

    bool used = wsum[kk] > 0.0f;
    out[(size_t)kk * 256 + dcol] =
        used ? ((1.0f - cfg::EMA_M) * hold + cfg::EMA_M * nh) : hold;
  }
}

// ---------------------------------------------------------------------------
extern "C" void kernel_launch(void* const* d_in, const int* in_sizes, int n_in,
                              void* d_out, int out_size, void* d_ws, size_t ws_size,
                              hipStream_t stream) {
  using namespace cfg;
  const float* hf     = (const float*)d_in[0];
  const float* hs     = (const float*)d_in[1];
  const int*   idx    = (const int*)d_in[2];
  const float* wgt    = (const float*)d_in[3];
  const float* states = (const float*)d_in[4];
  const float* W_m    = (const float*)d_in[5];
  const float* w_ih   = (const float*)d_in[6];
  const float* w_hh   = (const float*)d_in[7];
  const float* b_ih   = (const float*)d_in[8];
  const float* b_hh   = (const float*)d_in[9];
  float* out = (float*)d_out;

  // workspace carve-up (256B aligned)
  char* ws = (char*)d_ws;
  size_t off = 0;
  auto carve = [&](size_t bytes) {
    char* p = ws + off;
    off = (off + bytes + 255) & ~(size_t)255;
    return p;
  };
  float*    m_f32   = (float*)carve((size_t)BATCH * DIM * 4);        // 64 MB
  int*      counts  = (int*)carve(KTOT * 4);                         // counts+cursor
  int*      cursor  = (int*)carve(KTOT * 4);                         //   zeroed together
  int*      offsets = (int*)carve((KTOT + 1) * 4);
  int*      entry_b = (int*)carve(BATCH * KROUTE * 4);               // 1 MB
  float*    entry_w = (float*)carve(BATCH * KROUTE * 4);             // 1 MB
  float*    wsum    = (float*)carve(KTOT * 4);
  unsigned* wm_bf   = (unsigned*)carve(DIM * INDIM * 2);             // 256 KB
  unsigned* wih_bf  = (unsigned*)carve(3 * DIM * DIM * 2);           // 384 KB
  unsigned* whh_bf  = (unsigned*)carve(3 * DIM * DIM * 2);           // 384 KB
  unsigned* am_bf   = (unsigned*)carve((size_t)KTOT * DIM * 2);      // 4 MB
  unsigned* st_bf   = (unsigned*)carve((size_t)KTOT * DIM * 2);      // 4 MB (~75 MB total)

  // 1) counting sort of contributions (int atomics only)
  k_zero_i32<<<(2 * KTOT + 255) / 256, 256, 0, stream>>>(counts, 2 * KTOT);
  k_count<<<(BATCH * KROUTE + 255) / 256, 256, 0, stream>>>(idx, counts);
  k_scan<<<1, 256, 0, stream>>>(counts, offsets);
  k_fill<<<(BATCH * KROUTE + 255) / 256, 256, 0, stream>>>(idx, wgt, offsets,
                                                           cursor, entry_b, entry_w);
  // 2) pack weights / states to bf16
  k_cvt<<<(DIM * INDIM / 2 + 255) / 256, 256, 0, stream>>>(W_m, wm_bf, DIM * INDIM / 2);
  k_cvt<<<(3 * DIM * DIM / 2 + 255) / 256, 256, 0, stream>>>(w_ih, wih_bf, 3 * DIM * DIM / 2);
  k_cvt<<<(3 * DIM * DIM / 2 + 255) / 256, 256, 0, stream>>>(w_hh, whh_bf, 3 * DIM * DIM / 2);
  k_cvt<<<(KTOT * DIM / 2 + 255) / 256, 256, 0, stream>>>(states, st_bf, KTOT * DIM / 2);
  // 3) big GEMM (WMMA bf16) -> m
  k_gemm1<<<(BATCH / 16) * 2, 256, 0, stream>>>(hf, hs, wm_bf, m_f32);
  // 4) bucketed weighted gather -> agg_mean (bf16) + wsum
  k_gather<<<KTOT / 8, 256, 0, stream>>>(m_f32, offsets, entry_b, entry_w, wsum, am_bf);
  // 5) fused GRU + EMA blend (WMMA bf16)
  k_gru<<<(KTOT / 16) * 2, 256, 0, stream>>>(am_bf, st_bf, wih_bf, whh_bf,
                                             b_ih, b_hh, states, wsum, out);
}
